// Actor_attf_66975720013850
// MI455X (gfx1250) — matrix operand order repaired and checked by reading, not compiled, not run on hardware
//
#include <hip/hip_runtime.h>
#include <math.h>

// ---------------------------------------------------------------------------
// Actor attention network, fused single kernel for MI455X (gfx1250, wave32).
// A wave processes groups of 16 items so that EVERY matmul maps to WMMA with
// M = 16 (items or tokens):
//   Phase A: batched self MLP  [16it,4]@[4,32]@[32,16]  (f32x4 + f16x32 WMMA)
//   Phase B: per-item token encoders (others 15tok / food 16tok) + attention
//            + layernorm; obs staged via global_load_async_to_lds_b32
//   Phase C: batched head      [16it,48]@[48,32]@[32,32] (f16x32 WMMA), then
//            32->2 + tanh scalar (one (item,j) per lane), coalesced store.
// Inter-phase relayouts go through bank-padded LDS guarded by s_wait_dscnt.
// ---------------------------------------------------------------------------

typedef float    v2f  __attribute__((ext_vector_type(2)));
typedef float    v8f  __attribute__((ext_vector_type(8)));
typedef _Float16 v16h __attribute__((ext_vector_type(16)));

#define LDS_WAIT()   asm volatile("s_wait_dscnt 0" ::: "memory")
#define ASYNC_WAIT() asm volatile("s_wait_asynccnt 0" ::: "memory")

constexpr int OBSD   = 96;   // 4 + 15*4 + 16*2
constexpr int BLOCK  = 128;  // 4 waves
constexpr int WAVES  = 4;
constexpr int GROUPS = 4;    // groups of 16 items per wave
constexpr int HS     = 33;   // padded hidden row stride (bank-conflict free)
constexpr int SS     = 17;   // padded self_out row stride
constexpr int MSTR   = 49;   // padded m row stride

// LDS weight arena offsets (floats)
constexpr int EW1=0,    EB1=128,  EW2=160,  EB2=672,
              OW1=688,  OB1=816,  OW2=848,  OB2=1360, OG=1376,  OBt=1392,
              GW1=1408, GB1=1472, GW2=1504, GB2=2016, GG=2032,  GBt=2048,
              AW1=2064, AB1=3600, AW2=3632, AB2=4656, AW3=4688, AB3=4752,
              SWTOT=4754;
// per-wave scratch: obs[96] | hbuf[16*33] | sbuf[16*17] | mbuf[16*49]
constexpr int SCR_OBS=0, SCR_H=96, SCR_S=624, SCR_M=896, SCRSZ=1680;

__device__ __forceinline__ v8f wmma_f32_4(v2f a, v2f b, v8f c) {
  return __builtin_amdgcn_wmma_f32_16x16x4_f32(false, a, false, b, (short)0, c,
                                               false, false);
}
__device__ __forceinline__ v8f wmma_f32_32h(v16h a, v16h b, v8f c) {
  return __builtin_amdgcn_wmma_f32_16x16x32_f16(false, a, false, b, (short)0, c,
                                                false, false);
}

__device__ __forceinline__ float red16(float p) {   // sum over a 16-lane half
  p += __shfl_xor(p, 1, 32);
  p += __shfl_xor(p, 2, 32);
  p += __shfl_xor(p, 4, 32);
  p += __shfl_xor(p, 8, 32);
  return p;
}

__device__ __forceinline__ void cp(float* d, const float* s, int n, int tid) {
  for (int i = tid; i < n; i += BLOCK) d[i] = s[i];
}

// Gather a 16-bit A operand (ISA 7.12.2 16-bit A 16x32 layout) from an LDS
// matrix buf[row][stride]; K beyond kmax reads as zero (K padding).
__device__ __forceinline__ v16h gatherA16(const float* buf, int stride, int row,
                                          int hi, int koff, int kmax) {
  v16h a;
#pragma unroll
  for (int e = 0; e < 16; ++e) {
    int v = e >> 1, h = e & 1;
    int K = ((v < 4) ? ((hi ? 8 : 0) + 2 * v + h)
                     : ((hi ? 24 : 16) + 2 * (v - 4) + h)) + koff;
    a[e] = (K < kmax) ? (_Float16)buf[row * stride + K] : (_Float16)0.f;
  }
  return a;
}

// Build a 16-bit B operand (halves e -> K in A-table order, lane -> N) from row-major w[K][n].
__device__ __forceinline__ v16h buildB16(const float* w, int ncols, int n,
                                         int hi, int koff, int kmax) {
  v16h b;
#pragma unroll
  for (int e = 0; e < 16; ++e) {
    int K = ((e < 8) ? ((hi ? 8 : 0) + e) : ((hi ? 24 : 16) + (e - 8))) + koff;
    b[e] = (K < kmax) ? (_Float16)w[K * ncols + n] : (_Float16)0.f;
  }
  return b;
}

// Token encoder (WMMA 2-layer MLP) + single-query attention + layernorm+relu.
// Returns attended feature f (replicated across half-waves).
template <bool OTHERS>
__device__ __forceinline__ float encode_attend_ln(
    const float* obsS, float* hbuf, int hi, int f,
    v2f bt0, v2f bt1, const float* b1, v16h bw2, const float* b2,
    const float* lng, const float* lnb, float q) {
  // A (16x4 f32): lane = token (M), half-wave selects K pair
  v2f a;
  if (OTHERS) {
    float msk = (f < 15) ? 1.f : 0.f;          // pad token 15 with zeros
    int o = hi ? 34 : 4;                        // K(2,3) come from og[30+2t..]
    a.x = msk * obsS[o + 2 * f];
    a.y = msk * obsS[o + 1 + 2 * f];
  } else {
    a.x = hi ? 0.f : obsS[64 + 2 * f];          // food: K=2,3 zero-pad
    a.y = hi ? 0.f : obsS[65 + 2 * f];
  }
  v8f cz = {};
  v8f d0 = wmma_f32_4(a, bt0, cz);              // hidden cols 0..15
  v8f d1 = wmma_f32_4(a, bt1, cz);              // hidden cols 16..31
#pragma unroll
  for (int r = 0; r < 8; ++r) {                 // bias+relu -> LDS [tok][hid]
    int tok = r + (hi ? 8 : 0);
    hbuf[tok * HS + f]      = fmaxf(d0[r] + b1[f], 0.f);
    hbuf[tok * HS + 16 + f] = fmaxf(d1[r] + b1[16 + f], 0.f);
  }
  LDS_WAIT();
  v16h a16 = gatherA16(hbuf, HS, f, hi, 0, 32);
  v8f de = wmma_f32_32h(a16, bw2, cz);
  float enc[8];
#pragma unroll
  for (int r = 0; r < 8; ++r) enc[r] = fmaxf(de[r] + b2[f], 0.f);

  // attention scores: dot(enc[tok,:], q) via half-wave reductions
  float sc[16];
#pragma unroll
  for (int r = 0; r < 8; ++r) {
    float sown = red16(enc[r] * q);             // token r + 8*hi
    float soth = __shfl_xor(sown, 16, 32);      // token r + 8*(1-hi)
    sc[r + (hi ? 8 : 0)] = sown * 0.25f;        // / sqrt(NT=16)
    sc[r + (hi ? 0 : 8)] = soth * 0.25f;
  }
  if (OTHERS) sc[15] = -1e30f;                  // mask padded token
  float mx = sc[0];
#pragma unroll
  for (int t = 1; t < 16; ++t) mx = fmaxf(mx, sc[t]);
  float wgt[16], wsum = 0.f;
#pragma unroll
  for (int t = 0; t < 16; ++t) { wgt[t] = expf(sc[t] - mx); wsum += wgt[t]; }
  float inv = 1.f / wsum;
  float part = 0.f;
#pragma unroll
  for (int r = 0; r < 8; ++r) part += wgt[r + (hi ? 8 : 0)] * enc[r];
  part *= inv;
  float outv = part + __shfl_xor(part, 16, 32); // combine token halves

  // layernorm over the 16 features + relu
  float mu  = red16(outv) * (1.f / 16.f);
  float dv  = outv - mu;
  float var = red16(dv * dv) * (1.f / 16.f);
  return fmaxf(dv / sqrtf(var + 1e-5f) * lng[f] + lnb[f], 0.f);
}

__global__ __launch_bounds__(BLOCK) void actor_attf_kernel(
    const float* __restrict__ s_input,
    const float* en_w1, const float* en_b1, const float* en_w2, const float* en_b2,
    const float* oa_w1, const float* oa_b1, const float* oa_w2, const float* oa_b2,
    const float* oa_ln_g, const float* oa_ln_b,
    const float* g_w1,  const float* g_b1,  const float* g_w2,  const float* g_b2,
    const float* g_ln_g, const float* g_ln_b,
    const float* a_w1, const float* a_b1, const float* a_w2, const float* a_b2,
    const float* a_w3, const float* a_b3,
    float* __restrict__ out) {
  __shared__ float sw[SWTOT];
  __shared__ float sscr[WAVES][SCRSZ];
  const int tid = threadIdx.x;

  cp(sw + EW1, en_w1, 128, tid);  cp(sw + EB1, en_b1, 32, tid);
  cp(sw + EW2, en_w2, 512, tid);  cp(sw + EB2, en_b2, 16, tid);
  cp(sw + OW1, oa_w1, 128, tid);  cp(sw + OB1, oa_b1, 32, tid);
  cp(sw + OW2, oa_w2, 512, tid);  cp(sw + OB2, oa_b2, 16, tid);
  cp(sw + OG,  oa_ln_g, 16, tid); cp(sw + OBt, oa_ln_b, 16, tid);
  cp(sw + GW1, g_w1, 64, tid);    cp(sw + GB1, g_b1, 32, tid);
  cp(sw + GW2, g_w2, 512, tid);   cp(sw + GB2, g_b2, 16, tid);
  cp(sw + GG,  g_ln_g, 16, tid);  cp(sw + GBt, g_ln_b, 16, tid);
  cp(sw + AW1, a_w1, 1536, tid);  cp(sw + AB1, a_b1, 32, tid);
  cp(sw + AW2, a_w2, 1024, tid);  cp(sw + AB2, a_b2, 32, tid);
  cp(sw + AW3, a_w3, 64, tid);    cp(sw + AB3, a_b3, 2, tid);
  __syncthreads();

  const int wave = tid >> 5, L = tid & 31, hi = L >> 4, f = L & 15;
  float* scr  = sscr[wave];
  float* obsS = scr + SCR_OBS;
  float* hbuf = scr + SCR_H;
  float* sbuf = scr + SCR_S;   // self_out [item][16] stride SS
  float* mbuf = scr + SCR_M;   // m        [item][48] stride MSTR

  // ---- per-wave persistent WMMA B operands ----
  const int k0 = hi ? 2 : 0;                 // f32 B 4x16: halves hold K pairs
  v2f be1_t0, be1_t1, b_oa_t0, b_oa_t1, b_g_t0, b_g_t1;
  be1_t0.x = sw[EW1 + k0*32 + f];       be1_t0.y = sw[EW1 + (k0+1)*32 + f];
  be1_t1.x = sw[EW1 + k0*32 + 16 + f];  be1_t1.y = sw[EW1 + (k0+1)*32 + 16 + f];
  b_oa_t0.x = sw[OW1 + k0*32 + f];      b_oa_t0.y = sw[OW1 + (k0+1)*32 + f];
  b_oa_t1.x = sw[OW1 + k0*32 + 16 + f]; b_oa_t1.y = sw[OW1 + (k0+1)*32 + 16 + f];
  b_g_t0.x = hi ? 0.f : sw[GW1 + f];       b_g_t0.y = hi ? 0.f : sw[GW1 + 32 + f];
  b_g_t1.x = hi ? 0.f : sw[GW1 + 16 + f];  b_g_t1.y = hi ? 0.f : sw[GW1 + 32 + 16 + f];
  const v16h bew2   = buildB16(sw + EW2, 16, f, hi, 0, 32);
  const v16h bw2_oa = buildB16(sw + OW2, 16, f, hi, 0, 32);
  const v16h bw2_g  = buildB16(sw + GW2, 16, f, hi, 0, 32);
  const v16h bh1_00 = buildB16(sw + AW1, 32, f,      hi, 0,  48);
  const v16h bh1_01 = buildB16(sw + AW1, 32, 16 + f, hi, 0,  48);
  const v16h bh1_10 = buildB16(sw + AW1, 32, f,      hi, 32, 48);
  const v16h bh1_11 = buildB16(sw + AW1, 32, 16 + f, hi, 32, 48);
  const v16h bh2_0  = buildB16(sw + AW2, 32, f,      hi, 0,  32);
  const v16h bh2_1  = buildB16(sw + AW2, 32, 16 + f, hi, 0,  32);

  const unsigned obs_lds = (unsigned)(uintptr_t)obsS + 4u * (unsigned)L;
  const size_t wave_gid = (size_t)blockIdx.x * WAVES + wave;

  for (int grp = 0; grp < GROUPS; ++grp) {
    const size_t base = (wave_gid * GROUPS + grp) * 16;   // 16 items per group
    const float* gb = s_input + base * (size_t)OBSD;
    v8f cz = {};

    // ============ Phase A: batched self MLP (M = 16 items) ============
    v2f ax;
    ax.x = gb[f * OBSD + k0];
    ax.y = gb[f * OBSD + k0 + 1];
    v8f sd0 = wmma_f32_4(ax, be1_t0, cz);
    v8f sd1 = wmma_f32_4(ax, be1_t1, cz);
#pragma unroll
    for (int r = 0; r < 8; ++r) {
      int it = r + (hi ? 8 : 0);
      hbuf[it * HS + f]      = fmaxf(sd0[r] + sw[EB1 + f], 0.f);
      hbuf[it * HS + 16 + f] = fmaxf(sd1[r] + sw[EB1 + 16 + f], 0.f);
    }
    LDS_WAIT();
    v16h sa16 = gatherA16(hbuf, HS, f, hi, 0, 32);
    v8f sde = wmma_f32_32h(sa16, bew2, cz);
#pragma unroll
    for (int r = 0; r < 8; ++r) {
      int it = r + (hi ? 8 : 0);
      float so = fmaxf(sde[r] + sw[EB2 + f], 0.f);
      sbuf[it * SS + f]   = so;    // q source for phase B
      mbuf[it * MSTR + f] = so;    // m[0:16] = self_out
    }
    LDS_WAIT();

    // ============ Phase B: per-item encoders + attention ============
    for (int it = 0; it < 16; ++it) {
      const float* g = s_input + (base + it) * (size_t)OBSD;
      if (it + 1 < 16) __builtin_prefetch(g + OBSD, 0, 0);
      // async-stage 96 floats of obs into LDS (3 x B32 per lane)
      unsigned long long ga = (unsigned long long)(uintptr_t)g + 4ull * (unsigned)L;
      asm volatile("global_load_async_to_lds_b32 %0, %1, off"
                   :: "v"(obs_lds), "v"(ga) : "memory");
      asm volatile("global_load_async_to_lds_b32 %0, %1, off offset:128"
                   :: "v"(obs_lds), "v"(ga) : "memory");
      asm volatile("global_load_async_to_lds_b32 %0, %1, off offset:256"
                   :: "v"(obs_lds), "v"(ga) : "memory");
      ASYNC_WAIT();

      const float q = sbuf[it * SS + f];
      const float other_out = encode_attend_ln<true>(
          obsS, hbuf, hi, f, b_oa_t0, b_oa_t1, sw + OB1, bw2_oa, sw + OB2,
          sw + OG, sw + OBt, q);
      const float food_out = encode_attend_ln<false>(
          obsS, hbuf, hi, f, b_g_t0, b_g_t1, sw + GB1, bw2_g, sw + GB2,
          sw + GG, sw + GBt, q);
      if (L < 16) {
        mbuf[it * MSTR + 16 + f] = food_out;
        mbuf[it * MSTR + 32 + f] = other_out;
      }
      LDS_WAIT();
    }

    // ============ Phase C: batched head (M = 16 items) ============
    // GEMM1: [16,48] @ a_w1[48,32], K padded to 64 (two f16 K-tiles)
    v16h am0 = gatherA16(mbuf, MSTR, f, hi, 0, 48);
    v16h am1 = gatherA16(mbuf, MSTR, f, hi, 32, 48);
    v8f e0 = wmma_f32_32h(am0, bh1_00, cz);  e0 = wmma_f32_32h(am1, bh1_10, e0);
    v8f e1 = wmma_f32_32h(am0, bh1_01, cz);  e1 = wmma_f32_32h(am1, bh1_11, e1);
#pragma unroll
    for (int r = 0; r < 8; ++r) {            // leaky_relu(0.01) -> LDS
      int it = r + (hi ? 8 : 0);
      float x0 = e0[r] + sw[AB1 + f];
      float x1 = e1[r] + sw[AB1 + 16 + f];
      hbuf[it * HS + f]      = (x0 > 0.f) ? x0 : 0.01f * x0;
      hbuf[it * HS + 16 + f] = (x1 > 0.f) ? x1 : 0.01f * x1;
    }
    LDS_WAIT();
    // GEMM2: [16,32] @ a_w2[32,32]
    v16h ah = gatherA16(hbuf, HS, f, hi, 0, 32);
    v8f f0 = wmma_f32_32h(ah, bh2_0, cz);
    v8f f1 = wmma_f32_32h(ah, bh2_1, cz);
#pragma unroll
    for (int r = 0; r < 8; ++r) {
      int it = r + (hi ? 8 : 0);
      float x0 = f0[r] + sw[AB2 + f];
      float x1 = f1[r] + sw[AB2 + 16 + f];
      hbuf[it * HS + f]      = (x0 > 0.f) ? x0 : 0.01f * x0;
      hbuf[it * HS + 16 + f] = (x1 > 0.f) ? x1 : 0.01f * x1;
    }
    LDS_WAIT();
    // out[item, j] = tanh(h2 @ a_w3 + a_b3): lane -> (item=f, j=hi)
    float o = sw[AB3 + hi];
#pragma unroll
    for (int k = 0; k < 32; ++k) o += hbuf[f * HS + k] * sw[AW3 + k * 2 + hi];
    out[(base + f) * 2 + hi] = tanhf(o);
    LDS_WAIT();
  }
}

extern "C" void kernel_launch(void* const* d_in, const int* in_sizes, int n_in,
                              void* d_out, int out_size, void* d_ws, size_t ws_size,
                              hipStream_t stream) {
  const float* s_input = (const float*)d_in[0];
  const float* en_w1 = (const float*)d_in[1];
  const float* en_b1 = (const float*)d_in[2];
  const float* en_w2 = (const float*)d_in[3];
  const float* en_b2 = (const float*)d_in[4];
  const float* oa_w1 = (const float*)d_in[5];
  const float* oa_b1 = (const float*)d_in[6];
  const float* oa_w2 = (const float*)d_in[7];
  const float* oa_b2 = (const float*)d_in[8];
  const float* oa_ln_g = (const float*)d_in[9];
  const float* oa_ln_b = (const float*)d_in[10];
  const float* g_w1 = (const float*)d_in[11];
  const float* g_b1 = (const float*)d_in[12];
  const float* g_w2 = (const float*)d_in[13];
  const float* g_b2 = (const float*)d_in[14];
  const float* g_ln_g = (const float*)d_in[15];
  const float* g_ln_b = (const float*)d_in[16];
  const float* a_w1 = (const float*)d_in[17];
  const float* a_b1 = (const float*)d_in[18];
  const float* a_w2 = (const float*)d_in[19];
  const float* a_b2 = (const float*)d_in[20];
  const float* a_w3 = (const float*)d_in[21];
  const float* a_b3 = (const float*)d_in[22];
  float* out = (float*)d_out;

  const long n_items = (long)in_sizes[0] / OBSD;                 // 262144
  const int blocks = (int)(n_items / (WAVES * GROUPS * 16));     // 1024 (exact)
  actor_attf_kernel<<<blocks, BLOCK, 0, stream>>>(
      s_input, en_w1, en_b1, en_w2, en_b2, oa_w1, oa_b1, oa_w2, oa_b2,
      oa_ln_g, oa_ln_b, g_w1, g_b1, g_w2, g_b2, g_ln_g, g_ln_b,
      a_w1, a_b1, a_w2, a_b2, a_w3, a_b3, out);
}
